// Attn_Net_Gated_71786083385800
// MI455X (gfx1250) — matrix operand order repaired and checked by reading, not compiled, not run on hardware
//
#include <hip/hip_runtime.h>
#include <hip/hip_bf16.h>

typedef __attribute__((ext_vector_type(16))) _Float16 v16h;
typedef __attribute__((ext_vector_type(8)))  _Float16 v8h;
typedef __attribute__((ext_vector_type(4)))  _Float16 v4h;
typedef __attribute__((ext_vector_type(8)))  float    v8f;

#define THREADS 256
#define DFIX 1024
#define LFIX 128

// ---------------------------------------------------------------------------
// Shared epilogue: gate + project + per-row reduction to scalar logit.
// D-matrix layout: VGPR r = row (r + 8*half), col = l16 (+ 16*wave).
// ---------------------------------------------------------------------------
__device__ __forceinline__ void gated_epilogue(
    const v8f& accA, const v8f& accB,
    int wave, int half, int l16, int rowBase,
    const float* __restrict__ ba, const float* __restrict__ bb,
    const float* __restrict__ Wc, const float* __restrict__ bc,
    float* __restrict__ s_out)
{
    const int   col   = wave * 16 + l16;
    const float wc    = Wc[col];
    const float biasA = ba[col];
    const float biasB = bb[col];

    __shared__ float sPart[8][16];
    float part[8];
#pragma unroll
    for (int r = 0; r < 8; ++r) {
        const float av  = accA[r] + biasA;
        const float bv  = accB[r] + biasB;
        const float sig = 1.0f / (1.0f + __expf(-av));
        float g = sig * tanhf(bv) * wc;
#pragma unroll
        for (int m = 1; m < 16; m <<= 1) g += __shfl_xor(g, m, 32);
        part[r] = g;
    }
    if (l16 == 0) {      // lane 0 -> rows 0..7, lane 16 -> rows 8..15
#pragma unroll
        for (int r = 0; r < 8; ++r) sPart[wave][half * 8 + r] = part[r];
    }
    __syncthreads();
    if (threadIdx.x < 16) {
        float s = bc[0];
#pragma unroll
        for (int w = 0; w < 8; ++w) s += sPart[w][threadIdx.x];
        s_out[rowBase + threadIdx.x] = s;   // TEMP == 1.0
    }
}

// ---------------------------------------------------------------------------
// Prologue: convert Wa/Wb to f16 once (2 x L x D = 512 KB).  These stay
// resident in the 192 MB L2 for all GEMM blocks.
// ---------------------------------------------------------------------------
__global__ __launch_bounds__(THREADS)
void cvt_weights_kernel(const float* __restrict__ Wa, const float* __restrict__ Wb,
                        _Float16* __restrict__ WaH, _Float16* __restrict__ WbH,
                        int n)
{
    const int i = blockIdx.x * THREADS + threadIdx.x;
    if (i < n) {
        WaH[i] = (_Float16)Wa[i];
        WbH[i] = (_Float16)Wb[i];
    }
}

// ---------------------------------------------------------------------------
// FUSED fast path (D=1024, L=128): one block = 16 feature rows.
//
// Phase 1: L2-normalize the 16 rows.  Row r is owned by the 16 lanes of one
//   half-wave (thread = r*16+sub); sum-of-squares reduced with __shfl_xor,
//   row values kept in registers, fp32 result stored to the feature output
//   and an f16 copy deposited directly into a 32 KB LDS tile.
//
// Phase 2: dual WMMA GEMM over all 128 output columns.  8 waves; wave w owns
//   col tile [16w,16w+16) with both Wa/Wb accumulators.  A-fragments come
//   from LDS (ds_load_b128, shared across waves); B-fragments are f16 weights
//   streamed from L2.  Fragment addressing (contiguous K, same order for A and B):
//     A: lane l -> row l%16, K = khalf*16+[0..15]
//     B: lane n -> weight row n%16, K = khalf*16+[0..15]
// ---------------------------------------------------------------------------
__global__ __launch_bounds__(256)
void fused_norm_gemm_kernel(const float* __restrict__ feat,
                            float* __restrict__ featN,
                            const _Float16* __restrict__ WaH,
                            const _Float16* __restrict__ WbH,
                            const float* __restrict__ ba, const float* __restrict__ bb,
                            const float* __restrict__ Wc, const float* __restrict__ bc,
                            float* __restrict__ s_out)
{
    __shared__ _Float16 tileA[16][DFIX];     // 32 KB f16 A-tile

    const int tid     = threadIdx.x;
    const int rowBase = blockIdx.x * 16;

    // ---------------- Phase 1: normalize 16 rows ----------------
    {
        const int r   = tid >> 4;            // row within tile (0..15)
        const int sub = tid & 15;            // 16 threads per row
        const float4* src = (const float4*)(feat  + (size_t)(rowBase + r) * DFIX);
        float4*       dst = (float4*)(featN + (size_t)(rowBase + r) * DFIX);

        float4 v[16];
        float ss = 0.0f;
#pragma unroll
        for (int j = 0; j < 16; ++j) {
            v[j] = src[j * 16 + sub];        // coalesced across the 16 lanes
            ss += v[j].x * v[j].x + v[j].y * v[j].y
                + v[j].z * v[j].z + v[j].w * v[j].w;
        }
        // reduce across the 16 lanes owning this row (stays in one half-wave)
#pragma unroll
        for (int m = 1; m < 16; m <<= 1) ss += __shfl_xor(ss, m, 32);
        const float inv = 1.0f / fmaxf(sqrtf(ss), 1e-12f);

        v4h* ldsRow = (v4h*)&tileA[r][0];
#pragma unroll
        for (int j = 0; j < 16; ++j) {
            float4 w = v[j];
            w.x *= inv; w.y *= inv; w.z *= inv; w.w *= inv;
            dst[j * 16 + sub] = w;           // fp32 normalized feature (output)
            v4h h;
            h[0] = (_Float16)w.x; h[1] = (_Float16)w.y;
            h[2] = (_Float16)w.z; h[3] = (_Float16)w.w;
            ldsRow[j * 16 + sub] = h;        // f16 A-tile into LDS
        }
    }
    __syncthreads();

    // ---------------- Phase 2: dual WMMA GEMM ----------------
    const int wave = tid >> 5;               // 0..7 -> column tile
    const int lane = tid & 31;
    const int half = lane >> 4;              // K-half selector
    const int l16  = lane & 15;

    const _Float16* aLds  = &tileA[l16][half * 16];
    const _Float16* waPtr = WaH + (size_t)(wave * 16 + l16) * DFIX + half * 16;
    const _Float16* wbPtr = WbH + (size_t)(wave * 16 + l16) * DFIX + half * 16;

    v8f accA = {};
    v8f accB = {};

#pragma unroll 4
    for (int k = 0; k < DFIX; k += 32) {
        const v8h a0 = *(const v8h*)(aLds + k);        // ds_load_b128
        const v8h a1 = *(const v8h*)(aLds + k + 8);    // ds_load_b128
        const v16h af = __builtin_shufflevector(
            a0, a1, 0, 1, 2, 3, 4, 5, 6, 7, 8, 9, 10, 11, 12, 13, 14, 15);
        const v16h bfa = *(const v16h*)(waPtr + k);    // 2 x global_load_b128
        const v16h bfb = *(const v16h*)(wbPtr + k);    // 2 x global_load_b128

        accA = __builtin_amdgcn_wmma_f32_16x16x32_f16(
            false, af, false, bfa, (short)0, accA, false, false);
        accB = __builtin_amdgcn_wmma_f32_16x16x32_f16(
            false, af, false, bfb, (short)0, accB, false, false);
    }

    gated_epilogue(accA, accB, wave, half, l16, rowBase, ba, bb, Wc, bc, s_out);
}

// ---------------------------------------------------------------------------
// Generic fallback path (any D % 32 == 0, L == 128): separate normalize +
// fp32-operand GEMM with in-loop conversion.
// ---------------------------------------------------------------------------
__global__ __launch_bounds__(THREADS)
void normalize_kernel(const float* __restrict__ feat, float* __restrict__ featN,
                      int Dcols)
{
    const size_t row = blockIdx.x;
    const float4* src = (const float4*)(feat + row * (size_t)Dcols);
    float4*       dst = (float4*)(featN + row * (size_t)Dcols);
    const int nvec = Dcols >> 2;

    __shared__ float red[THREADS];
    float ss = 0.0f;
    for (int i = threadIdx.x; i < nvec; i += THREADS) {
        float4 v = src[i];
        ss += v.x * v.x + v.y * v.y + v.z * v.z + v.w * v.w;
    }
    red[threadIdx.x] = ss;
    __syncthreads();
    for (int s = THREADS >> 1; s > 0; s >>= 1) {
        if (threadIdx.x < s) red[threadIdx.x] += red[threadIdx.x + s];
        __syncthreads();
    }
    const float inv = 1.0f / fmaxf(sqrtf(red[0]), 1e-12f);
    for (int i = threadIdx.x; i < nvec; i += THREADS) {
        float4 v = src[i];
        v.x *= inv; v.y *= inv; v.z *= inv; v.w *= inv;
        dst[i] = v;
    }
}

__device__ __forceinline__ v16h load16_f16(const float* __restrict__ p)
{
    const float4 x0 = ((const float4*)p)[0];
    const float4 x1 = ((const float4*)p)[1];
    const float4 x2 = ((const float4*)p)[2];
    const float4 x3 = ((const float4*)p)[3];
    v16h r;
    r[0]  = (_Float16)x0.x; r[1]  = (_Float16)x0.y; r[2]  = (_Float16)x0.z; r[3]  = (_Float16)x0.w;
    r[4]  = (_Float16)x1.x; r[5]  = (_Float16)x1.y; r[6]  = (_Float16)x1.z; r[7]  = (_Float16)x1.w;
    r[8]  = (_Float16)x2.x; r[9]  = (_Float16)x2.y; r[10] = (_Float16)x2.z; r[11] = (_Float16)x2.w;
    r[12] = (_Float16)x3.x; r[13] = (_Float16)x3.y; r[14] = (_Float16)x3.z; r[15] = (_Float16)x3.w;
    return r;
}

__device__ __forceinline__ v16h load8x2_f16(const float* __restrict__ p0,
                                            const float* __restrict__ p1)
{
    const float4 a0 = ((const float4*)p0)[0];
    const float4 a1 = ((const float4*)p0)[1];
    const float4 b0 = ((const float4*)p1)[0];
    const float4 b1 = ((const float4*)p1)[1];
    v16h r;
    r[0]  = (_Float16)a0.x; r[1]  = (_Float16)a0.y; r[2]  = (_Float16)a0.z; r[3]  = (_Float16)a0.w;
    r[4]  = (_Float16)a1.x; r[5]  = (_Float16)a1.y; r[6]  = (_Float16)a1.z; r[7]  = (_Float16)a1.w;
    r[8]  = (_Float16)b0.x; r[9]  = (_Float16)b0.y; r[10] = (_Float16)b0.z; r[11] = (_Float16)b0.w;
    r[12] = (_Float16)b1.x; r[13] = (_Float16)b1.y; r[14] = (_Float16)b1.z; r[15] = (_Float16)b1.w;
    return r;
}

__global__ __launch_bounds__(256)
void gated_score_f32_kernel(const float* __restrict__ featN,
                            const float* __restrict__ Wa, const float* __restrict__ ba,
                            const float* __restrict__ Wb, const float* __restrict__ bb,
                            const float* __restrict__ Wc, const float* __restrict__ bc,
                            float* __restrict__ s_out, int Dcols)
{
    const int rowBase = blockIdx.x * 16;
    const int wave = threadIdx.x >> 5;
    const int lane = threadIdx.x & 31;
    const int half = lane >> 4;
    const int l16  = lane & 15;

    const float* aPtr  = featN + (size_t)(rowBase + l16) * Dcols;
    const float* waPtr = Wa + (size_t)(wave * 16 + l16) * Dcols;
    const float* wbPtr = Wb + (size_t)(wave * 16 + l16) * Dcols;

    v8f accA = {};
    v8f accB = {};

    for (int k = 0; k < Dcols; k += 32) {
        const v16h af = load8x2_f16(aPtr + k + half * 16,
                                    aPtr + k + half * 16 + 8);
        const v16h bfa = load16_f16(waPtr + k + half * 16);
        const v16h bfb = load16_f16(wbPtr + k + half * 16);
        accA = __builtin_amdgcn_wmma_f32_16x16x32_f16(
            false, af, false, bfa, (short)0, accA, false, false);
        accB = __builtin_amdgcn_wmma_f32_16x16x32_f16(
            false, af, false, bfb, (short)0, accB, false, false);
    }

    gated_epilogue(accA, accB, wave, half, l16, rowBase, ba, bb, Wc, bc, s_out);
}

// ---------------------------------------------------------------------------
// Binary search helper on the sorted batch-id array.
// ---------------------------------------------------------------------------
__device__ __forceinline__ int lower_bound_i32(const int* __restrict__ batch,
                                               int n, int key)
{
    int lo = 0, hi = n;
    while (lo < hi) {
        const int mid = (lo + hi) >> 1;
        if (batch[mid] < key) lo = mid + 1; else hi = mid;
    }
    return lo;
}

// ---------------------------------------------------------------------------
// Kernel 3: per-segment softmax (one block per bag).  Deterministic tree
// reductions for max and sum; no atomics.
// ---------------------------------------------------------------------------
__global__ __launch_bounds__(THREADS)
void seg_softmax_kernel(const int* __restrict__ batch, const float* __restrict__ s,
                        float* __restrict__ score, int Nrows)
{
    const int b     = blockIdx.x;
    const int start = lower_bound_i32(batch, Nrows, b);
    const int end   = lower_bound_i32(batch, Nrows, b + 1);
    const int t     = threadIdx.x;

    __shared__ float red[THREADS];

    float mx = -3.402823466e38f;
    for (int i = start + t; i < end; i += THREADS) mx = fmaxf(mx, s[i]);
    red[t] = mx;
    __syncthreads();
    for (int k = THREADS >> 1; k > 0; k >>= 1) {
        if (t < k) red[t] = fmaxf(red[t], red[t + k]);
        __syncthreads();
    }
    mx = red[0];
    __syncthreads();

    float sum = 0.0f;
    for (int i = start + t; i < end; i += THREADS) sum += __expf(s[i] - mx);
    red[t] = sum;
    __syncthreads();
    for (int k = THREADS >> 1; k > 0; k >>= 1) {
        if (t < k) red[t] += red[t + k];
        __syncthreads();
    }
    const float inv = 1.0f / red[0];

    for (int i = start + t; i < end; i += THREADS)
        score[i] = __expf(s[i] - mx) * inv;
}

// ---------------------------------------------------------------------------
// Kernel 4: segment-sum of score * normalized-feature -> out (B x D).
// One block per (segment, 256-col chunk); fixed-order accumulation.
// ---------------------------------------------------------------------------
__global__ __launch_bounds__(THREADS)
void seg_pool_kernel(const int* __restrict__ batch, const float* __restrict__ score,
                     const float* __restrict__ featN, float* __restrict__ out,
                     int Nrows, int Dcols)
{
    const int b   = blockIdx.y;
    const int col = blockIdx.x * THREADS + threadIdx.x;
    if (col >= Dcols) return;

    const int start = lower_bound_i32(batch, Nrows, b);
    const int end   = lower_bound_i32(batch, Nrows, b + 1);

    float acc = 0.0f;
    for (int i = start; i < end; ++i)
        acc += score[i] * featN[(size_t)i * Dcols + col];
    out[(size_t)b * Dcols + col] = acc;
}

// ---------------------------------------------------------------------------
// Host launcher.
// Inputs:  feature(NxD f32), batch(N i32, sorted), Wa(LxD), ba(L), Wb(LxD),
//          bb(L), Wc(1xL), bc(1)
// Output:  [ out(BxD) | score(N) | featureN(NxD) ] flat f32
// Scratch: [ logits: N f32 | WaH: L*D f16 | WbH: L*D f16 ]  (~1.5 MB)
// ---------------------------------------------------------------------------
extern "C" void kernel_launch(void* const* d_in, const int* in_sizes, int n_in,
                              void* d_out, int out_size, void* d_ws, size_t ws_size,
                              hipStream_t stream)
{
    const float* feat  = (const float*)d_in[0];
    const int*   batch = (const int*)d_in[1];
    const float* Wa    = (const float*)d_in[2];
    const float* ba    = (const float*)d_in[3];
    const float* Wb    = (const float*)d_in[4];
    const float* bb    = (const float*)d_in[5];
    const float* Wc    = (const float*)d_in[6];
    const float* bc    = (const float*)d_in[7];

    const int N = in_sizes[1];
    const int D = in_sizes[0] / N;                       // 1024
    const int L = in_sizes[3];                           // 128
    const long long B = ((long long)out_size - (long long)N
                         - (long long)N * D) / D;        // 64

    float* outPool  = (float*)d_out;                     // B x D
    float* outScore = outPool + (size_t)B * D;           // N
    float* outFeat  = outScore + (size_t)N;              // N x D

    float*    sLogit = (float*)d_ws;                     // N f32
    _Float16* WaH    = (_Float16*)(sLogit + N);          // L*D f16
    _Float16* WbH    = WaH + (size_t)L * D;              // L*D f16

    const size_t needFast = (size_t)N * 4 + (size_t)2 * L * D * 2;
    const bool fast = (D == DFIX) && (L == LFIX) && (N % 16 == 0)
                   && (ws_size >= needFast);

    if (fast) {
        const int nW = L * D;
        cvt_weights_kernel<<<(nW + THREADS - 1) / THREADS, THREADS, 0, stream>>>(
            Wa, Wb, WaH, WbH, nW);
        fused_norm_gemm_kernel<<<N / 16, 256, 0, stream>>>(
            feat, outFeat, WaH, WbH, ba, bb, Wc, bc, sLogit);
    } else {
        normalize_kernel<<<N, THREADS, 0, stream>>>(feat, outFeat, D);
        gated_score_f32_kernel<<<N / 16, 256, 0, stream>>>(
            outFeat, Wa, ba, Wb, bb, Wc, bc, sLogit, D);
    }

    seg_softmax_kernel<<<(int)B, THREADS, 0, stream>>>(batch, sLogit, outScore, N);

    dim3 poolGrid((D + THREADS - 1) / THREADS, (unsigned)B);
    seg_pool_kernel<<<poolGrid, THREADS, 0, stream>>>(
        batch, outScore, outFeat, outPool, N, D);
}
